// GATSpatialEncoder_46084999086506
// MI455X (gfx1250) — matrix operand order repaired and checked by reading, not compiled, not run on hardware
//
#include <hip/hip_runtime.h>
#include <hip/hip_bf16.h>

// ---- problem constants (from setup_inputs) ----
#define B_   8
#define N_   1000
#define T_   12
#define F_   8
#define D_   64
#define H_   8
#define Cc_  8
#define L_   2
#define E_   16000
#define G_   (B_ * T_)        // 96
#define ET_  (E_ + N_)        // 17000 edges incl. self loops
#define MROWS (G_ * N_)       // 96000 rows for the 64x64 GEMMs

typedef __attribute__((ext_vector_type(16))) _Float16 v16h;
typedef __attribute__((ext_vector_type(8)))  float    v8f;
typedef __attribute__((ext_vector_type(4)))  float    f32x4;  // POD 16-byte vector (union-safe)

union AV { v16h v; f32x4 f[2]; };

// ---------------------------------------------------------------------------
// Pack Wl/Wr (f32, row-major KxN, both layers) into the CDNA5 WMMA B-matrix
// lane/VGPR layout for 16x16x32 f16:
//   lane = n + 16*((k>>3)&1), VGPR j holds packed K pair (kin, kin+1), kin = 8*(lane>>4) + 2*(j&3) + 16*(j>>2).
// bpack dword order: [layer][mat][ntile][kchunk][lane][vgpr]  (8192 dwords total)
// ---------------------------------------------------------------------------
__global__ void k_pack_b(const float* __restrict__ Wl, const float* __restrict__ Wr,
                         _Float16* __restrict__ bpack) {
    int idx = blockIdx.x * blockDim.x + threadIdx.x;
    if (idx >= 8192) return;
    int vgpr  = idx & 7;
    int lane  = (idx >> 3) & 31;
    int kc    = (idx >> 8) & 1;
    int nt    = (idx >> 9) & 3;
    int mat   = (idx >> 11) & 1;
    int layer = (idx >> 12) & 1;
    int n   = lane & 15;
    int kin = ((lane >> 4) << 3) + (vgpr & 3) * 2 + ((vgpr >> 2) << 4);   // K within 32-wide chunk
    int k   = kc * 32 + kin;
    int col = nt * 16 + n;
    const float* W = (mat == 0 ? Wl : Wr) + layer * D_ * D_;
    bpack[idx * 2 + 0] = (_Float16)W[(k + 0) * D_ + col];
    bpack[idx * 2 + 1] = (_Float16)W[(k + 1) * D_ + col];
}

// ---------------------------------------------------------------------------
// Input projection: h[g= b*T+t, n, d] = x[b,n,t,:] @ in_w + in_b   (K = 8)
// Writes f32 (residual/LN path) and f16 (WMMA A operand).
// ---------------------------------------------------------------------------
__global__ void k_inproj(const float* __restrict__ x, const float* __restrict__ in_w,
                         const float* __restrict__ in_b,
                         float* __restrict__ h32, _Float16* __restrict__ h16) {
    int i = blockIdx.x * blockDim.x + threadIdx.x;
    if (i >= MROWS * D_) return;
    int d   = i & 63;
    int row = i >> 6;              // g*N + n
    int g = row / N_, n = row % N_;
    int b = g / T_,  t = g % T_;
    const float* xp = x + ((size_t)(b * N_ + n) * T_ + t) * F_;
    float acc = in_b[d];
#pragma unroll
    for (int f = 0; f < F_; ++f) acc += xp[f] * in_w[f * D_ + d];
    h32[i] = acc;
    h16[i] = (_Float16)acc;
}

// ---------------------------------------------------------------------------
// WMMA GEMM: xl = h@Wl + bl, xr = h@Wr + br.  One wave per 16-row tile.
// 96000 rows = 6000 tiles = 750 blocks * 8 waves exactly -> no divergence,
// EXEC all-ones as WMMA requires. A layout (16x32 f16): lane L (half=L>>4,
// r=L&15) holds row m0+r, K = kbase + half*8 + [0..7] (v0-3) and
// K = kbase + 16 + half*8 + [0..7] (v4-7) -> two contiguous 16-B loads.
// ---------------------------------------------------------------------------
__global__ void __launch_bounds__(256)
k_gemm(const _Float16* __restrict__ h16, const _Float16* __restrict__ bpack,
       const float* __restrict__ bl, const float* __restrict__ br,
       float* __restrict__ xl, float* __restrict__ xr) {
    const int lane = threadIdx.x & 31;
    const int wave = threadIdx.x >> 5;
    const int tile = blockIdx.x * 8 + wave;     // < 6000 exactly
    const int m0   = tile * 16;
    const int half = lane >> 4;
    const int r    = lane & 15;

    AV a0, a1;
    const _Float16* rowp = h16 + (size_t)(m0 + r) * D_;
    a0.f[0] = *(const f32x4*)(rowp +      half * 8);
    a0.f[1] = *(const f32x4*)(rowp + 16 + half * 8);
    a1.f[0] = *(const f32x4*)(rowp + 32 + half * 8);
    a1.f[1] = *(const f32x4*)(rowp + 48 + half * 8);

    const int orow = m0 + half * 8;
    const int ncol = lane & 15;

#pragma unroll
    for (int nt = 0; nt < 4; ++nt) {
        // ---- Wl path ----
        AV b0, b1;
        const _Float16* p0 = bpack + ((((0 * 4 + nt) * 2 + 0) * 32 + lane) * 16);
        const _Float16* p1 = bpack + ((((0 * 4 + nt) * 2 + 1) * 32 + lane) * 16);
        b0.f[0] = *(const f32x4*)p0;  b0.f[1] = *(const f32x4*)(p0 + 8);
        b1.f[0] = *(const f32x4*)p1;  b1.f[1] = *(const f32x4*)(p1 + 8);
        v8f c = {};
        c = __builtin_amdgcn_wmma_f32_16x16x32_f16(false, a0.v, false, b0.v, (short)0, c, false, false);
        c = __builtin_amdgcn_wmma_f32_16x16x32_f16(false, a1.v, false, b1.v, (short)0, c, false, false);
        float biasl = bl[nt * 16 + ncol];
#pragma unroll
        for (int i = 0; i < 8; ++i)
            xl[(size_t)(orow + i) * D_ + nt * 16 + ncol] = c[i] + biasl;

        // ---- Wr path ----
        const _Float16* q0 = bpack + ((((1 * 4 + nt) * 2 + 0) * 32 + lane) * 16);
        const _Float16* q1 = bpack + ((((1 * 4 + nt) * 2 + 1) * 32 + lane) * 16);
        b0.f[0] = *(const f32x4*)q0;  b0.f[1] = *(const f32x4*)(q0 + 8);
        b1.f[0] = *(const f32x4*)q1;  b1.f[1] = *(const f32x4*)(q1 + 8);
        v8f d = {};
        d = __builtin_amdgcn_wmma_f32_16x16x32_f16(false, a0.v, false, b0.v, (short)0, d, false, false);
        d = __builtin_amdgcn_wmma_f32_16x16x32_f16(false, a1.v, false, b1.v, (short)0, d, false, false);
        float biasr = br[nt * 16 + ncol];
#pragma unroll
        for (int i = 0; i < 8; ++i)
            xr[(size_t)(orow + i) * D_ + nt * 16 + ncol] = d[i] + biasr;
    }
}

// ---------------------------------------------------------------------------
__global__ void k_init(float* __restrict__ nmax, float* __restrict__ nsum,
                       float* __restrict__ acc) {
    int i = blockIdx.x * blockDim.x + threadIdx.x;
    if (i >= MROWS * D_) return;
    acc[i] = 0.f;
    if (i < G_ * N_ * H_) { nmax[i] = -INFINITY; nsum[i] = 0.f; }
}

__device__ inline void atomicMaxF(float* addr, float val) {
    if (val >= 0.f) atomicMax((int*)addr, __float_as_int(val));
    else            atomicMin((unsigned int*)addr, __float_as_uint(val));
}

// e = sum_c leaky_relu(xl[src]+xr[dst]) * att ; atomic segment max over dst
__global__ void k_edge_logit(const float* __restrict__ xl, const float* __restrict__ xr,
                             const int* __restrict__ ei, const float* __restrict__ att,
                             float* __restrict__ ew, float* __restrict__ nmax) {
    int i = blockIdx.x * blockDim.x + threadIdx.x;
    if (i >= G_ * ET_ * H_) return;
    int h = i & 7;
    int e = (i >> 3) % ET_;
    int g = (i >> 3) / ET_;
    int s, d;
    if (e < E_) { s = ei[e]; d = ei[E_ + e]; } else { s = d = e - E_; }
    const float* pl = xl + (size_t)(g * N_ + s) * D_ + h * Cc_;
    const float* pr = xr + (size_t)(g * N_ + d) * D_ + h * Cc_;
    const float* pa = att + h * Cc_;
    float acc = 0.f;
#pragma unroll
    for (int c = 0; c < Cc_; ++c) {
        float v = pl[c] + pr[c];
        v = (v >= 0.f) ? v : 0.2f * v;
        acc += v * pa[c];
    }
    ew[i] = acc;
    atomicMaxF(&nmax[(g * N_ + d) * H_ + h], acc);
}

// ex = exp(e - max[dst]); atomic segment sum
__global__ void k_edge_exp(const int* __restrict__ ei, float* __restrict__ ew,
                           const float* __restrict__ nmax, float* __restrict__ nsum) {
    int i = blockIdx.x * blockDim.x + threadIdx.x;
    if (i >= G_ * ET_ * H_) return;
    int h = i & 7;
    int e = (i >> 3) % ET_;
    int g = (i >> 3) / ET_;
    int d = (e < E_) ? ei[E_ + e] : (e - E_);
    float m = nmax[(g * N_ + d) * H_ + h];
    if (m == -INFINITY) m = 0.f;           // mirrors jnp.where(isfinite(m), m, 0)
    float ex = __expf(ew[i] - m);
    ew[i] = ex;
    atomicAdd(&nsum[(g * N_ + d) * H_ + h], ex);
}

// msg scatter: acc[dst] += (ex/sum[dst]) * xl[src]
__global__ void k_edge_scatter(const int* __restrict__ ei, const float* __restrict__ ew,
                               const float* __restrict__ nsum, const float* __restrict__ xl,
                               float* __restrict__ acc) {
    int i = blockIdx.x * blockDim.x + threadIdx.x;
    if (i >= G_ * ET_ * H_) return;
    int h = i & 7;
    int e = (i >> 3) % ET_;
    int g = (i >> 3) / ET_;
    int s, d;
    if (e < E_) { s = ei[e]; d = ei[E_ + e]; } else { s = d = e - E_; }
    float alpha = ew[i] / nsum[(g * N_ + d) * H_ + h];
    const float* pl = xl + (size_t)(g * N_ + s) * D_ + h * Cc_;
    float* pd = acc + (size_t)(g * N_ + d) * D_ + h * Cc_;
#pragma unroll
    for (int c = 0; c < Cc_; ++c) atomicAdd(&pd[c], alpha * pl[c]);
}

// ELU + residual + LayerNorm(D=64) per node row; one wave32 per row,
// lane handles d and d+32, allreduce via shuffle-xor. Regenerates f16 h.
__global__ void __launch_bounds__(256)
k_finalize(const float* __restrict__ acc, const float* __restrict__ bias,
           const float* __restrict__ lng, const float* __restrict__ lnb,
           float* __restrict__ h32, _Float16* __restrict__ h16) {
    int lane = threadIdx.x & 31;
    int row  = blockIdx.x * 8 + (threadIdx.x >> 5);
    if (row >= MROWS) return;
    size_t base = (size_t)row * D_;
    int d0 = lane, d1 = lane + 32;
    float t0 = acc[base + d0] + bias[d0];
    float t1 = acc[base + d1] + bias[d1];
    t0 = (t0 > 0.f) ? t0 : (__expf(t0) - 1.f);   // ELU
    t1 = (t1 > 0.f) ? t1 : (__expf(t1) - 1.f);
    t0 += h32[base + d0];                        // residual
    t1 += h32[base + d1];
    float s = t0 + t1;
#pragma unroll
    for (int o = 16; o > 0; o >>= 1) s += __shfl_xor(s, o, 32);
    float mu = s * (1.f / 64.f);
    float q = (t0 - mu) * (t0 - mu) + (t1 - mu) * (t1 - mu);
#pragma unroll
    for (int o = 16; o > 0; o >>= 1) q += __shfl_xor(q, o, 32);
    float rstd = rsqrtf(q * (1.f / 64.f) + 1e-5f);
    float o0 = (t0 - mu) * rstd * lng[d0] + lnb[d0];
    float o1 = (t1 - mu) * rstd * lng[d1] + lnb[d1];
    h32[base + d0] = o0;  h16[base + d0] = (_Float16)o0;
    h32[base + d1] = o1;  h16[base + d1] = (_Float16)o1;
}

// (G,N,D) with g=b*T+t  ->  out (B,N,T,D)
__global__ void k_out(const float* __restrict__ h32, float* __restrict__ out) {
    int i = blockIdx.x * blockDim.x + threadIdx.x;
    if (i >= B_ * N_ * T_ * D_) return;
    int d = i & 63;
    int t = (i >> 6) % T_;
    int n = ((i >> 6) / T_) % N_;
    int b = (i >> 6) / (T_ * N_);
    out[i] = h32[(size_t)((b * T_ + t) * N_ + n) * D_ + d];
}

// ---------------------------------------------------------------------------
extern "C" void kernel_launch(void* const* d_in, const int* in_sizes, int n_in,
                              void* d_out, int out_size, void* d_ws, size_t ws_size,
                              hipStream_t stream) {
    (void)in_sizes; (void)n_in; (void)out_size; (void)ws_size;
    const float* x    = (const float*)d_in[0];
    const int*   ei   = (const int*)  d_in[1];
    const float* in_w = (const float*)d_in[2];
    const float* in_b = (const float*)d_in[3];
    const float* Wl   = (const float*)d_in[4];
    const float* bWl  = (const float*)d_in[5];
    const float* Wr   = (const float*)d_in[6];
    const float* bWr  = (const float*)d_in[7];
    const float* att  = (const float*)d_in[8];
    const float* bias = (const float*)d_in[9];
    const float* lng  = (const float*)d_in[10];
    const float* lnb  = (const float*)d_in[11];
    float* out = (float*)d_out;

    // workspace carve-up (256-B aligned slabs)
    char* ws = (char*)d_ws;
    size_t off = 0;
    auto carve = [&](size_t bytes) { char* p = ws + off; off += (bytes + 255) & ~(size_t)255; return p; };
    float*    h32   = (float*)   carve((size_t)MROWS * D_ * 4);        // 24.6 MB
    _Float16* h16   = (_Float16*)carve((size_t)MROWS * D_ * 2);        // 12.3 MB
    float*    xl    = (float*)   carve((size_t)MROWS * D_ * 4);
    float*    xr    = (float*)   carve((size_t)MROWS * D_ * 4);
    float*    ew    = (float*)   carve((size_t)G_ * ET_ * H_ * 4);     // 52.2 MB
    float*    nmax  = (float*)   carve((size_t)G_ * N_ * H_ * 4);
    float*    nsum  = (float*)   carve((size_t)G_ * N_ * H_ * 4);
    float*    accb  = (float*)   carve((size_t)MROWS * D_ * 4);
    _Float16* bpack = (_Float16*)carve(8192 * 4);                      // 32 KB

    const int TPB = 256;
    const int gElem  = (MROWS * D_ + TPB - 1) / TPB;       // 24000
    const int gEdge  = (G_ * ET_ * H_ + TPB - 1) / TPB;    // 51000
    const int gGemm  = (MROWS / 16) / 8;                   // 750 (exact)
    const int gNode  = MROWS / 8;                          // 12000 (exact)

    k_pack_b <<<32, TPB, 0, stream>>>(Wl, Wr, bpack);
    k_inproj <<<gElem, TPB, 0, stream>>>(x, in_w, in_b, h32, h16);

    for (int l = 0; l < L_; ++l) {
        const _Float16* bp = bpack + (size_t)l * 8192;     // per-layer slab (halves)
        k_init        <<<gElem, TPB, 0, stream>>>(nmax, nsum, accb);
        k_gemm        <<<gGemm, TPB, 0, stream>>>(h16, bp, bWl + l * D_, bWr + l * D_, xl, xr);
        k_edge_logit  <<<gEdge, TPB, 0, stream>>>(xl, xr, ei, att + l * H_ * Cc_, ew, nmax);
        k_edge_exp    <<<gEdge, TPB, 0, stream>>>(ei, ew, nmax, nsum);
        k_edge_scatter<<<gEdge, TPB, 0, stream>>>(ei, ew, nsum, xl, accb);
        k_finalize    <<<gNode, TPB, 0, stream>>>(accb, bias + l * D_, lng + l * D_, lnb + l * D_, h32, h16);
    }

    k_out <<<gElem, TPB, 0, stream>>>(h32, out);
}
